// MultiHeadAttention_88330297410289
// MI455X (gfx1250) — matrix operand order repaired and checked by reading, not compiled, not run on hardware
//
#include <hip/hip_runtime.h>
#include <hip/hip_bf16.h>

// ---------------------------------------------------------------------------
// MHA forward for MI455X (gfx1250, wave32, WMMA bf16 path + async LDS staging)
// B=32, S=256, H=2048, nh=16, hd=128
// ---------------------------------------------------------------------------

typedef __bf16 bf16_t;
typedef __attribute__((ext_vector_type(16))) __bf16 v16bf;
typedef __attribute__((ext_vector_type(8)))  float  v8f;

#define HD    128
#define NH    16
#define SEQ   256
#define HID   2048
#define BATCH 32
#define MROWS (BATCH * SEQ)   // 8192

union ABFrag { v16bf v; uint4 u[2]; };

__device__ __forceinline__ v8f wmma_bf16(v16bf a, v16bf b, v8f c) {
    // 8 args: (neg_a, A, neg_b, B, c_mod, C, reuse_a, reuse_b)
    return __builtin_amdgcn_wmma_f32_16x16x32_bf16(
        false, a, false, b, (short)0, c, false, false);
}

// Async global->LDS copy of 32 bytes per lane (2 x b128), ASYNCcnt-tracked.
// ISA 08: the instruction offset is added to both the LDS and the global address
__device__ __forceinline__ void async_copy32(bf16_t* lds_dst, const bf16_t* gsrc) {
    unsigned la = (unsigned)(unsigned long long)(size_t)lds_dst; // addr[31:0] = LDS addr
    unsigned long long ga = (unsigned long long)(size_t)gsrc;
    asm volatile("global_load_async_to_lds_b128 %0, %1, off\n\t"
                 "global_load_async_to_lds_b128 %0, %1, off offset:16"
                 :: "v"(la), "v"(ga) : "memory");
}

__device__ __forceinline__ void wait_async0() {
    asm volatile("s_wait_asynccnt 0" ::: "memory");
}

// ---------------------------------------------------------------------------
// fp32 -> bf16 conversion
// ---------------------------------------------------------------------------
__global__ void cvt_bf16_kernel(const float* __restrict__ src,
                                bf16_t* __restrict__ dst, int n) {
    int i = blockIdx.x * blockDim.x + threadIdx.x;
    int stride = gridDim.x * blockDim.x;
    for (; i < n; i += stride) dst[i] = (bf16_t)src[i];
}

// ---------------------------------------------------------------------------
// C[M,N] = A[M,K] @ B[N,K]^T   (A, B bf16 row-major over K)
// OUT_MODE 0: fp32 row-major [M,N]
// OUT_MODE 1: bf16 [b, h, s, d]      (Q/K pre-RoPE layout)
// OUT_MODE 2: bf16 [b, h, d, s]      (V transposed for attention B-fragments)
// Block: 256 threads = 8 waves; BM=128, BN=128, BK=32.
// Double-buffered LDS, async global->LDS staging overlapped with WMMA.
// ---------------------------------------------------------------------------
template <int OUT_MODE>
__global__ __launch_bounds__(256)
void gemm_bf16_kernel(const bf16_t* __restrict__ A,
                      const bf16_t* __restrict__ B,
                      void* __restrict__ Cout,
                      int M, int N, int K) {
    __shared__ __align__(16) bf16_t As[2][128 * 32];   // 2 x 8 KB
    __shared__ __align__(16) bf16_t Bs[2][128 * 32];   // 2 x 8 KB

    const int tid  = threadIdx.x;
    const int lane = tid & 31;
    const int w    = tid >> 5;
    const int wm   = w >> 1;          // 0..3
    const int wn   = w & 1;           // 0..1
    const int bn   = blockIdx.x * 128;
    const int bm   = blockIdx.y * 128;

    const int lrow = tid & 127;       // staging row
    const int lcol = (tid >> 7) * 16; // staging k-chunk {0,16}

    const int nl  = lane & 15;        // lane-in-half
    const int g   = lane >> 4;        // half select
    const int akb = g * 16;           // A fragment k-base (contiguous order: k = 16*g + s)
    const int bkb = g * 16;           // B fragment k-base (contiguous order: k = 16*g + s)

    const bf16_t* gA = A + (size_t)(bm + lrow) * K + lcol;
    const bf16_t* gB = B + (size_t)(bn + lrow) * K + lcol;

    v8f acc[2][4] = {};

    // Prologue: stage tile 0 into buffer 0
    async_copy32(&As[0][lrow * 32 + lcol], gA);
    async_copy32(&Bs[0][lrow * 32 + lcol], gB);
    wait_async0();
    __syncthreads();

    int cur = 0;
    for (int k0 = 0; k0 < K; k0 += 32) {
        const int nxt = cur ^ 1;
        if (k0 + 32 < K) {  // stage next tile while computing current
            async_copy32(&As[nxt][lrow * 32 + lcol], gA + k0 + 32);
            async_copy32(&Bs[nxt][lrow * 32 + lcol], gB + k0 + 32);
        }

        ABFrag af[2], bfr[4];
        #pragma unroll
        for (int i = 0; i < 2; ++i) {
            int r = wm * 32 + i * 16 + nl;   // A row
            af[i].u[0] = *(const uint4*)(&As[cur][r * 32 + akb]);
            af[i].u[1] = *(const uint4*)(&As[cur][r * 32 + akb + 8]);
        }
        #pragma unroll
        for (int j = 0; j < 4; ++j) {
            int r = wn * 64 + j * 16 + nl;   // B row (= output column)
            bfr[j].u[0] = *(const uint4*)(&Bs[cur][r * 32 + bkb]);
            bfr[j].u[1] = *(const uint4*)(&Bs[cur][r * 32 + bkb + 8]);
        }
        #pragma unroll
        for (int i = 0; i < 2; ++i)
            #pragma unroll
            for (int j = 0; j < 4; ++j)
                acc[i][j] = wmma_bf16(af[i].v, bfr[j].v, acc[i][j]);

        wait_async0();       // next tile landed in LDS (this wave's asyncs)
        __syncthreads();     // everyone done reading 'cur' + staging 'nxt'
        cur = nxt;
    }

    // Epilogue: C/D layout -> element (m = j + 8*(lane>=16), n = lane&15)
    const int mm = g * 8;
    #pragma unroll
    for (int i = 0; i < 2; ++i) {
        #pragma unroll
        for (int j = 0; j < 4; ++j) {
            int n = bn + wn * 64 + j * 16 + nl;
            #pragma unroll
            for (int r = 0; r < 8; ++r) {
                int m = bm + wm * 32 + i * 16 + r + mm;
                float v = acc[i][j][r];
                if (OUT_MODE == 0) {
                    ((float*)Cout)[(size_t)m * N + n] = v;
                } else {
                    int b = m >> 8, s = m & 255;      // S = 256
                    int h = n >> 7, d = n & 127;      // hd = 128
                    if (OUT_MODE == 1)
                        ((bf16_t*)Cout)[(((size_t)(b * NH + h)) * SEQ + s) * HD + d] = (bf16_t)v;
                    else
                        ((bf16_t*)Cout)[(((size_t)(b * NH + h)) * HD + d) * SEQ + s] = (bf16_t)v;
                }
            }
        }
    }
}

// ---------------------------------------------------------------------------
// In-place RoPE on Q and K, layout [b, h, s, d]; one thread owns pair (d,d+64)
// ---------------------------------------------------------------------------
__global__ void rope_kernel(bf16_t* __restrict__ Q, bf16_t* __restrict__ Kv,
                            int total) {
    int i = blockIdx.x * blockDim.x + threadIdx.x;
    if (i >= total) return;
    int d  = i & 63;
    int s  = (i >> 6) & 255;
    int bh = i >> 14;
    size_t base = ((size_t)bh * SEQ + s) * HD;
    // inv_freq = 10000^(-d/64) = exp(-ln(10000)/64 * d)
    float ang = (float)s * __expf((float)d * -0.14391156343f);
    float sn, cs;
    __sincosf(ang, &sn, &cs);

    float q1 = (float)Q[base + d], q2 = (float)Q[base + 64 + d];
    Q[base + d]      = (bf16_t)(q1 * cs - q2 * sn);
    Q[base + 64 + d] = (bf16_t)(q2 * cs + q1 * sn);

    float k1 = (float)Kv[base + d], k2 = (float)Kv[base + 64 + d];
    Kv[base + d]      = (bf16_t)(k1 * cs - k2 * sn);
    Kv[base + 64 + d] = (bf16_t)(k2 * cs + k1 * sn);
}

// ---------------------------------------------------------------------------
// Flash-style attention. 1 block per (b,h); 512 threads = 16 waves;
// each wave owns 16 query rows and streams key chunks of 64.
// Q,K: bf16 [bh, s, d];  Vt: bf16 [bh, d, s];  ctx out: bf16 [b, s, h*hd+d]
// ---------------------------------------------------------------------------
__global__ __launch_bounds__(512)
void attn_kernel(const bf16_t* __restrict__ Q, const bf16_t* __restrict__ Kmat,
                 const bf16_t* __restrict__ Vt, bf16_t* __restrict__ ctx) {
    __shared__ __align__(16) bf16_t plds[16 * 16 * 64];   // 32 KB, wave-private slices

    const int bh   = blockIdx.x;
    const int b    = bh >> 4, h = bh & 15;
    const int tid  = threadIdx.x;
    const int lane = tid & 31;
    const int w    = tid >> 5;
    const int qb   = w * 16;
    const int g    = lane >> 4;
    const int nl   = lane & 15;
    const float scale = 0.08838834764831845f; // 1/sqrt(128)

    // Load Q A-fragments once (4 k-tiles over hd=128)
    ABFrag qf[4];
    {
        size_t qrow = ((size_t)bh * SEQ + qb + nl) * HD;
        int akb = g * 16;
        #pragma unroll
        for (int kt = 0; kt < 4; ++kt) {
            qf[kt].u[0] = *(const uint4*)(Q + qrow + kt * 32 + akb);
            qf[kt].u[1] = *(const uint4*)(Q + qrow + kt * 32 + akb + 8);
        }
    }

    float mstat[8], lstat[8];
    #pragma unroll
    for (int j = 0; j < 8; ++j) { mstat[j] = -1e30f; lstat[j] = 0.0f; }
    v8f acc_o[8] = {};

    bf16_t* myp = plds + w * (16 * 64);

    for (int c0 = 0; c0 < SEQ; c0 += 64) {
        // --- scores: Q(16x128) @ K_chunk^T(128x64) ---
        v8f accs[4] = {};
        #pragma unroll
        for (int kt = 0; kt < 4; ++kt) {
            #pragma unroll
            for (int t = 0; t < 4; ++t) {
                ABFrag kf;
                size_t krow = ((size_t)bh * SEQ + c0 + t * 16 + nl) * HD
                              + kt * 32 + g * 16;
                kf.u[0] = *(const uint4*)(Kmat + krow);
                kf.u[1] = *(const uint4*)(Kmat + krow + 8);
                accs[t] = wmma_bf16(qf[kt].v, kf.v, accs[t]);
            }
        }

        // --- online softmax (row r lives at VGPR j, half g: r = j + 8g) ---
        float cmax[8];
        #pragma unroll
        for (int j = 0; j < 8; ++j) cmax[j] = -1e30f;
        #pragma unroll
        for (int t = 0; t < 4; ++t)
            #pragma unroll
            for (int j = 0; j < 8; ++j) {
                accs[t][j] *= scale;
                cmax[j] = fmaxf(cmax[j], accs[t][j]);
            }
        #pragma unroll
        for (int j = 0; j < 8; ++j)
            #pragma unroll
            for (int off = 8; off >= 1; off >>= 1)
                cmax[j] = fmaxf(cmax[j], __shfl_xor(cmax[j], off, 32));

        float alpha[8], rsum[8];
        #pragma unroll
        for (int j = 0; j < 8; ++j) {
            float mn = fmaxf(mstat[j], cmax[j]);
            alpha[j] = __expf(mstat[j] - mn);
            mstat[j] = mn;
            rsum[j]  = 0.0f;
        }
        #pragma unroll
        for (int t = 0; t < 4; ++t)
            #pragma unroll
            for (int j = 0; j < 8; ++j) {
                float p = __expf(accs[t][j] - mstat[j]);
                rsum[j] += p;
                myp[(j + 8 * g) * 64 + t * 16 + nl] = (bf16_t)p;  // C -> LDS
            }
        #pragma unroll
        for (int j = 0; j < 8; ++j)
            #pragma unroll
            for (int off = 8; off >= 1; off >>= 1)
                rsum[j] += __shfl_xor(rsum[j], off, 32);
        #pragma unroll
        for (int j = 0; j < 8; ++j) lstat[j] = lstat[j] * alpha[j] + rsum[j];
        #pragma unroll
        for (int dt = 0; dt < 8; ++dt)
            #pragma unroll
            for (int j = 0; j < 8; ++j) acc_o[dt][j] *= alpha[j];

        // --- re-swizzle P via wave-private LDS into A-fragments ---
        ABFrag pf[2];
        {
            int akb = g * 16;
            const bf16_t* prow = myp + nl * 64;
            #pragma unroll
            for (int kt2 = 0; kt2 < 2; ++kt2) {
                pf[kt2].u[0] = *(const uint4*)(prow + kt2 * 32 + akb);
                pf[kt2].u[1] = *(const uint4*)(prow + kt2 * 32 + akb + 8);
            }
        }

        // --- P(16x64) @ V_chunk(64x128): Vt is [bh, d, s] so B cols contiguous
        #pragma unroll
        for (int dt = 0; dt < 8; ++dt) {
            #pragma unroll
            for (int kt2 = 0; kt2 < 2; ++kt2) {
                ABFrag vf;
                size_t vrow = ((size_t)bh * HD + dt * 16 + nl) * SEQ
                              + c0 + kt2 * 32 + g * 16;
                vf.u[0] = *(const uint4*)(Vt + vrow);
                vf.u[1] = *(const uint4*)(Vt + vrow + 8);
                acc_o[dt] = wmma_bf16(pf[kt2].v, vf.v, acc_o[dt]);
            }
        }
    }

    // --- normalize, write ctx bf16 [b, s, h*hd + d] ---
    #pragma unroll
    for (int dt = 0; dt < 8; ++dt) {
        int d = h * HD + dt * 16 + nl;
        #pragma unroll
        for (int j = 0; j < 8; ++j) {
            int s  = qb + j + 8 * g;
            float v = acc_o[dt][j] / lstat[j];
            ctx[((size_t)(b * SEQ + s)) * HID + d] = (bf16_t)v;
        }
    }
}

// ---------------------------------------------------------------------------
// Launch
// ---------------------------------------------------------------------------
extern "C" void kernel_launch(void* const* d_in, const int* in_sizes, int n_in,
                              void* d_out, int out_size, void* d_ws, size_t ws_size,
                              hipStream_t stream) {
    const float* X  = (const float*)d_in[0];
    const float* Wq = (const float*)d_in[1];
    const float* Wk = (const float*)d_in[2];
    const float* Wv = (const float*)d_in[3];
    const float* Wo = (const float*)d_in[4];
    float* out = (float*)d_out;

    const size_t NX = (size_t)MROWS * HID;  // 16,777,216
    const size_t NW = (size_t)HID * HID;    //  4,194,304

    bf16_t* Xb  = (bf16_t*)d_ws;
    bf16_t* Wqb = Xb  + NX;
    bf16_t* Wkb = Wqb + NW;
    bf16_t* Wvb = Wkb + NW;
    bf16_t* Wob = Wvb + NW;
    bf16_t* Qr  = Wob + NW;   // [b,h,s,d]
    bf16_t* Kr  = Qr  + NX;   // [b,h,s,d]
    bf16_t* Vt  = Kr  + NX;   // [b,h,d,s]
    bf16_t* Ctx = Vt  + NX;   // [b,s,H]

    cvt_bf16_kernel<<<4096, 256, 0, stream>>>(X,  Xb,  (int)NX);
    cvt_bf16_kernel<<<2048, 256, 0, stream>>>(Wq, Wqb, (int)NW);
    cvt_bf16_kernel<<<2048, 256, 0, stream>>>(Wk, Wkb, (int)NW);
    cvt_bf16_kernel<<<2048, 256, 0, stream>>>(Wv, Wvb, (int)NW);
    cvt_bf16_kernel<<<2048, 256, 0, stream>>>(Wo, Wob, (int)NW);

    dim3 gg(HID / 128, MROWS / 128);  // (16, 64)
    gemm_bf16_kernel<1><<<gg, 256, 0, stream>>>(Xb, Wqb, Qr, MROWS, HID, HID);
    gemm_bf16_kernel<1><<<gg, 256, 0, stream>>>(Xb, Wkb, Kr, MROWS, HID, HID);
    gemm_bf16_kernel<2><<<gg, 256, 0, stream>>>(Xb, Wvb, Vt, MROWS, HID, HID);

    int total = BATCH * NH * SEQ * 64;  // 8,388,608 rotation pairs
    rope_kernel<<<total / 256, 256, 0, stream>>>(Qr, Kr, total);

    attn_kernel<<<BATCH * NH, 512, 0, stream>>>(Qr, Kr, Vt, Ctx);

    gemm_bf16_kernel<0><<<gg, 256, 0, stream>>>(Ctx, Wob, out, MROWS, HID, HID);
}
